// Decomposeable_76725295775815
// MI455X (gfx1250) — compile-verified, matrix operand order checked by reading, not run on hardware
//
#include <hip/hip_runtime.h>

// ---------------- CDNA5 WMMA / TDM types ----------------
typedef __attribute__((ext_vector_type(16))) __bf16 v16bf;
typedef __attribute__((ext_vector_type(8)))  float  v8f;
typedef __attribute__((ext_vector_type(4)))  unsigned int u32x4;
typedef __attribute__((ext_vector_type(8)))  int    i32x8;
typedef __attribute__((ext_vector_type(4)))  int    i32x4;

union FragB { v16bf v; unsigned int u[8]; };

#ifndef __has_builtin
#define __has_builtin(x) 0
#endif
#if __has_builtin(__builtin_amdgcn_tensor_load_to_lds)
#define HAVE_TDM 1
#else
#define HAVE_TDM 0
#endif

// ---------------------------------------------------------------------------
// Generic (optionally batched) bf16 GEMM:  C = A(MxK) * Bt(NxK)^T  [+bias][relu]
// A  : row-major [M, lda] bf16, K-contiguous, K % 32 == 0 (zero padded)
// Bt : row-major [Npad, ldb] bf16, Npad >= roundup(N,128) rows VALID (zero pad)
//      -> no masking on operand loads at all.
// Block tile 128x128 (8 waves as 4x2); each wave: 32x64 macro-tile,
// 2 A-frags x 4 B-frags -> 8 v_wmma per K-step.  REQUIRES M % 128 == 0.
// ---------------------------------------------------------------------------
__global__ __launch_bounds__(256)
void gemm_wmma(const __bf16* __restrict__ A, int lda, long sA,
               const __bf16* __restrict__ Bt, int ldb, long sB,
               float* __restrict__ C, int ldc, long sC,
               __bf16* __restrict__ Cb, int ldcb, long sCb,
               const float* __restrict__ bias,
               int M, int N, int K, int relu)
{
    const int bz = blockIdx.z;
    A  += (long)bz * sA;
    Bt += (long)bz * sB;

    const int wave = threadIdx.x >> 5;
    const int lane = threadIdx.x & 31;
    const int m0   = blockIdx.y * 128 + (wave >> 1) * 32;
    const int n0   = blockIdx.x * 128 + (wave & 1) * 64;
    const int hi   = lane >> 4;                 // half-wave K +8 swizzle
    const int l15  = lane & 15;

    const __bf16* arow0 = A + (long)(m0 + l15) * lda;
    const __bf16* arow1 = A + (long)(m0 + 16 + l15) * lda;
    const __bf16* brow[4];
#pragma unroll
    for (int j = 0; j < 4; ++j)
        brow[j] = Bt + (long)(n0 + j * 16 + l15) * ldb;

    v8f acc[2][4] = {};
    for (int k0 = 0; k0 < K; k0 += 32) {
        __builtin_prefetch(arow0 + k0 + 32, 0, 1);   // speculative, OOB-safe
        __builtin_prefetch(arow1 + k0 + 32, 0, 1);

        FragB fa[2], fb[4];
#pragma unroll
        for (int p = 0; p < 8; ++p) {
            // dword p holds bf16 pair at K = 2p (+8 for p>=4) (+8 for lanes 16-31)
            const int kk = k0 + 2 * p + ((p >= 4) ? 8 : 0) + hi * 8;
            fa[0].u[p] = *(const unsigned int*)(arow0 + kk);
            fa[1].u[p] = *(const unsigned int*)(arow1 + kk);
#pragma unroll
            for (int j = 0; j < 4; ++j)
                fb[j].u[p] = *(const unsigned int*)(brow[j] + kk);
        }
#pragma unroll
        for (int i = 0; i < 2; ++i)
#pragma unroll
            for (int j = 0; j < 4; ++j)
                acc[i][j] = __builtin_amdgcn_wmma_f32_16x16x32_bf16(
                                false, fa[i].v, false, fb[j].v,
                                (short)0, acc[i][j], false, false);
    }

#pragma unroll
    for (int j = 0; j < 4; ++j) {
        const int col = n0 + j * 16 + l15;
        const bool nv = col < N;
        const float bb = (bias != nullptr && nv) ? bias[col] : 0.0f;
#pragma unroll
        for (int i = 0; i < 2; ++i) {
#pragma unroll
            for (int r = 0; r < 8; ++r) {
                const int m = m0 + i * 16 + r + 8 * hi;
                float v = acc[i][j][r] + bb;
                if (relu) v = fmaxf(v, 0.0f);
                if (nv) {
                    if (C)  C [(long)bz * sC  + (long)m * ldc  + col] = v;
                    if (Cb) Cb[(long)bz * sCb + (long)m * ldcb + col] = (__bf16)v;
                }
            }
        }
    }
}

// ---------------------------------------------------------------------------
// K=224 GEMM with the whole Bt weight panel (256 x 224 bf16 = 114,688 B)
// staged in LDS by the Tensor Data Mover: wave 0 issues one tensor_load_to_lds
// (2-D D#: 4-byte elements, 112 dwords/row, 256 rows, contiguous), waits
// s_wait_tensorcnt 0, then all waves consume B fragments from LDS.
// ---------------------------------------------------------------------------
__global__ __launch_bounds__(256)
void gemm_wmma_wlds(const __bf16* __restrict__ A, int lda,
                    const __bf16* __restrict__ Bt,      // 256 x 224 panel
                    float* __restrict__ C, int ldc,
                    __bf16* __restrict__ Cb, int ldcb,
                    const float* __restrict__ bias,
                    int M, int N, int relu)
{
    __shared__ __bf16 bpan[256 * 224];

#if HAVE_TDM
    if (threadIdx.x < 32) {                      // wave 0 issues the DMA
        unsigned long long ga = (unsigned long long)(const void*)Bt;
        unsigned lds0 = (unsigned)(unsigned long long)(const void*)&bpan[0];
        u32x4 g0 = { 1u,                         // count=1, user descriptor
                     lds0,                       // D#.lds_addr
                     (unsigned)(ga & 0xFFFFFFFFull),
                     (unsigned)((ga >> 32) & 0x01FFFFFFull) | (2u << 30) }; // type=2
        i32x8 g1 = { (int)(2u << 16),            // data_size = 2 (4-byte units)
                     (int)(112u << 16),          // tensor_dim0 = 112 dwords/row
                     (int)(256u << 16),          // tensor_dim1 = 256 rows
                     (int)(112u << 16),          // tile_dim0   = 112
                     256,                        // tile_dim1   = 256
                     112,                        // tensor_dim0_stride = 112
                     0, 0 };
        i32x4 gz = { 0, 0, 0, 0 };
#if __clang_major__ >= 23
        i32x8 gz8 = { 0, 0, 0, 0, 0, 0, 0, 0 };
        __builtin_amdgcn_tensor_load_to_lds(g0, g1, gz, gz, gz8, 0);
#else
        __builtin_amdgcn_tensor_load_to_lds(g0, g1, gz, gz, 0);
#endif
        __builtin_amdgcn_s_wait_tensorcnt(0);
    }
    __syncthreads();
#else
    for (int i = threadIdx.x; i < 256 * 112; i += 256)
        ((unsigned int*)bpan)[i] = ((const unsigned int*)Bt)[i];
    __syncthreads();
#endif

    const int wave = threadIdx.x >> 5;
    const int lane = threadIdx.x & 31;
    const int m0   = blockIdx.y * 128 + (wave >> 1) * 32;
    const int n0   = blockIdx.x * 128 + (wave & 1) * 64;
    const int hi   = lane >> 4;
    const int l15  = lane & 15;

    const __bf16* arow0 = A + (long)(m0 + l15) * lda;
    const __bf16* arow1 = A + (long)(m0 + 16 + l15) * lda;
    int boff[4];
#pragma unroll
    for (int j = 0; j < 4; ++j)
        boff[j] = (n0 + j * 16 + l15) * 224;

    v8f acc[2][4] = {};
    for (int k0 = 0; k0 < 224; k0 += 32) {
        __builtin_prefetch(arow0 + k0 + 32, 0, 1);
        __builtin_prefetch(arow1 + k0 + 32, 0, 1);

        FragB fa[2], fb[4];
#pragma unroll
        for (int p = 0; p < 8; ++p) {
            const int kk = k0 + 2 * p + ((p >= 4) ? 8 : 0) + hi * 8;
            fa[0].u[p] = *(const unsigned int*)(arow0 + kk);
            fa[1].u[p] = *(const unsigned int*)(arow1 + kk);
#pragma unroll
            for (int j = 0; j < 4; ++j)
                fb[j].u[p] = *(const unsigned int*)&bpan[boff[j] + kk];
        }
#pragma unroll
        for (int i = 0; i < 2; ++i)
#pragma unroll
            for (int j = 0; j < 4; ++j)
                acc[i][j] = __builtin_amdgcn_wmma_f32_16x16x32_bf16(
                                false, fa[i].v, false, fb[j].v,
                                (short)0, acc[i][j], false, false);
    }

#pragma unroll
    for (int j = 0; j < 4; ++j) {
        const int col = n0 + j * 16 + l15;
        const bool nv = col < N;
        const float bb = (bias != nullptr && nv) ? bias[col] : 0.0f;
#pragma unroll
        for (int i = 0; i < 2; ++i) {
#pragma unroll
            for (int r = 0; r < 8; ++r) {
                const int m = m0 + i * 16 + r + 8 * hi;
                float v = acc[i][j][r] + bb;
                if (relu) v = fmaxf(v, 0.0f);
                if (nv) {
                    if (C)  C [(long)m * ldc  + col] = v;
                    if (Cb) Cb[(long)m * ldcb + col] = (__bf16)v;
                }
            }
        }
    }
}

// ---------------- elementwise / reduction kernels ----------------

__global__ void zero_bf(__bf16* p, long n) {
    long i = (long)blockIdx.x * blockDim.x + threadIdx.x;
    if (i < n) p[i] = (__bf16)0.0f;
}

// Wt[n*Kp + k] = W[k*Nsrc + n] for n < Nsrc, else 0.  Covers 256 padded rows.
__global__ void transpose_w(const float* __restrict__ W, __bf16* __restrict__ Wt,
                            int K, int Nsrc, int Kp) {
    long idx = (long)blockIdx.x * blockDim.x + threadIdx.x;
    if (idx >= 256L * Kp) return;
    int k = (int)(idx % Kp);
    int n = (int)(idx / Kp);
    float v = (k < K && n < Nsrc) ? W[(long)k * Nsrc + n] : 0.0f;
    Wt[idx] = (__bf16)v;
}

// gather embedding row, L2-normalize, store bf16 padded to 320 cols.
__global__ void gather_norm(const int* __restrict__ idx,
                            const float* __restrict__ emb,
                            __bf16* __restrict__ out, int ntok) {
    const int wave = threadIdx.x >> 5;
    const int lane = threadIdx.x & 31;
    const int tok = blockIdx.x * 8 + wave;
    if (tok >= ntok) return;
    const float* row = emb + (long)idx[tok] * 300;
    float ss = 0.0f;
    for (int k = lane; k < 300; k += 32) { float v = row[k]; ss += v * v; }
    for (int o = 16; o >= 1; o >>= 1) ss += __shfl_xor(ss, o);
    const float rn = rsqrtf(ss);
    __bf16* o = out + (long)tok * 320;
    for (int k = lane; k < 320; k += 32)
        o[k] = (k < 300) ? (__bf16)(row[k] * rn) : (__bf16)0.0f;
}

// sim[b,i,j] *= (i < len1[b]) * (j < len2[b])
__global__ void mask_sim(float* __restrict__ sim,
                         const int* __restrict__ len1,
                         const int* __restrict__ len2) {
    long idx = (long)blockIdx.x * blockDim.x + threadIdx.x;
    if (idx >= 128L * 256 * 256) return;
    int b = (int)(idx >> 16);
    int i = (int)((idx >> 8) & 255);
    int j = (int)(idx & 255);
    float m = (i < len1[b] && j < len2[b]) ? 1.0f : 0.0f;
    sim[idx] *= m;
}

// row softmax over last dim (256) -> bf16. One wave per row.
__global__ void softmax_rows(const float* __restrict__ sim,
                             __bf16* __restrict__ P, long nrows) {
    const int wave = threadIdx.x >> 5;
    const int lane = threadIdx.x & 31;
    const long row = (long)blockIdx.x * 8 + wave;
    if (row >= nrows) return;
    const float* s = sim + row * 256;
    float v[8], mx = -3.4e38f;
#pragma unroll
    for (int t = 0; t < 8; ++t) { v[t] = s[lane + 32 * t]; mx = fmaxf(mx, v[t]); }
    for (int o = 16; o >= 1; o >>= 1) mx = fmaxf(mx, __shfl_xor(mx, o));
    float sum = 0.0f;
#pragma unroll
    for (int t = 0; t < 8; ++t) { v[t] = __expf(v[t] - mx); sum += v[t]; }
    for (int o = 16; o >= 1; o >>= 1) sum += __shfl_xor(sum, o);
    const float inv = 1.0f / sum;
    __bf16* p = P + row * 256;
#pragma unroll
    for (int t = 0; t < 8; ++t) p[lane + 32 * t] = (__bf16)(v[t] * inv);
}

// column softmax: PT[b,j,i] = softmax_i(sim[b,i,j]). One wave per (b,j).
__global__ void softmax_cols(const float* __restrict__ sim,
                             __bf16* __restrict__ PT, long ncols) {
    const int wave = threadIdx.x >> 5;
    const int lane = threadIdx.x & 31;
    const long cidx = (long)blockIdx.x * 8 + wave;   // b*256 + j
    if (cidx >= ncols) return;
    const long b = cidx >> 8;
    const int  j = (int)(cidx & 255);
    const float* s = sim + b * 65536 + j;
    float v[8], mx = -3.4e38f;
#pragma unroll
    for (int t = 0; t < 8; ++t) { v[t] = s[(long)(lane + 32 * t) * 256]; mx = fmaxf(mx, v[t]); }
    for (int o = 16; o >= 1; o >>= 1) mx = fmaxf(mx, __shfl_xor(mx, o));
    float sum = 0.0f;
#pragma unroll
    for (int t = 0; t < 8; ++t) { v[t] = __expf(v[t] - mx); sum += v[t]; }
    for (int o = 16; o >= 1; o >>= 1) sum += __shfl_xor(sum, o);
    const float inv = 1.0f / sum;
    __bf16* p = PT + cidx * 256;
#pragma unroll
    for (int t = 0; t < 8; ++t) p[lane + 32 * t] = (__bf16)(v[t] * inv);
}

// xt[b, d, s] = x[(b*256+s)*200 + d] for d < 200, else 0.  256 padded rows.
__global__ void transpose_x(const float* __restrict__ x, __bf16* __restrict__ xt) {
    long idx = (long)blockIdx.x * blockDim.x + threadIdx.x;
    if (idx >= 128L * 256 * 256) return;
    int  s = (int)(idx & 255);
    long t = idx >> 8;
    int  d = (int)(t & 255);
    long b = t >> 8;
    xt[idx] = (d < 200) ? (__bf16)x[((long)b * 256 + s) * 200 + d] : (__bf16)0.0f;
}

// dst[m*ldd + n] = src[m*lds + n], n < N   (bf16 copy into concat slice)
__global__ void copy_slice_bf(const __bf16* __restrict__ src, int lds,
                              __bf16* __restrict__ dst, int ldd, long M, int N) {
    long idx = (long)blockIdx.x * blockDim.x + threadIdx.x;
    if (idx >= M * N) return;
    long m = idx / N;
    int  n = (int)(idx % N);
    dst[m * ldd + n] = src[m * lds + n];
}

// cath[b, 0:200] = sum_{s<len1[b]} v1[b,s,:];  cath[b, 200:400] = same for v2.
__global__ void aggregate(const float* __restrict__ v1, const float* __restrict__ v2,
                          const int* __restrict__ len1, const int* __restrict__ len2,
                          __bf16* __restrict__ cath) {
    int idx = blockIdx.x * blockDim.x + threadIdx.x;  // 128 * 400
    if (idx >= 128 * 400) return;
    int b = idx / 400;
    int c = idx % 400;
    const float* v = (c < 200) ? v1 : v2;
    const int    L = (c < 200) ? len1[b] : len2[b];
    const int    d = c % 200;
    const float* base = v + ((long)b * 256) * 200 + d;
    float s = 0.0f;
    for (int t = 0; t < L; ++t) s += base[(long)t * 200];
    cath[(long)b * 416 + c] = (__bf16)s;
}

// y[b,c] = sum_d h[b,d] * Wout[d,c] + bout[c]
__global__ void out_head(const float* __restrict__ h, const float* __restrict__ Wout,
                         const float* __restrict__ bout, float* __restrict__ y) {
    int idx = blockIdx.x * blockDim.x + threadIdx.x;
    if (idx >= 128 * 3) return;
    int b = idx / 3, c = idx % 3;
    float s = bout[c];
    const float* hb = h + (long)b * 200;
    for (int d = 0; d < 200; ++d) s += hb[d] * Wout[(long)d * 3 + c];
    y[idx] = s;
}

// ---------------------------------------------------------------------------
extern "C" void kernel_launch(void* const* d_in, const int* in_sizes, int n_in,
                              void* d_out, int out_size, void* d_ws, size_t ws_size,
                              hipStream_t stream) {
    (void)in_sizes; (void)n_in; (void)out_size; (void)ws_size;

    const int*   x1    = (const int*)d_in[0];
    const int*   x2    = (const int*)d_in[1];
    const int*   len1  = (const int*)d_in[2];
    const int*   len2  = (const int*)d_in[3];
    const float* emb   = (const float*)d_in[4];
    const float* W_proj= (const float*)d_in[5];
    const float* F_W1  = (const float*)d_in[6];
    const float* F_b1  = (const float*)d_in[7];
    const float* F_W2  = (const float*)d_in[8];
    const float* F_b2  = (const float*)d_in[9];
    const float* G_W1  = (const float*)d_in[10];
    const float* G_b1  = (const float*)d_in[11];
    const float* G_W2  = (const float*)d_in[12];
    const float* G_b2  = (const float*)d_in[13];
    const float* H_W1  = (const float*)d_in[14];
    const float* H_b1  = (const float*)d_in[15];
    const float* H_W2  = (const float*)d_in[16];
    const float* H_b2  = (const float*)d_in[17];
    const float* W_out = (const float*)d_in[18];
    const float* b_out = (const float*)d_in[19];
    float* y = (float*)d_out;
    char*  ws = (char*)d_ws;

    const long Bn = 128, Sn = 256, Dn = 200;
    const long M1 = Bn * Sn;                  // 32768 tokens
    const int  EP = 320, DP = 224, D2P = 416; // padded K dims (mult of 32)

    // ---------------- workspace layout ----------------
    const size_t SZ_E   = (size_t)M1 * EP * 2;
    const size_t OFF_E1 = 0;
    const size_t OFF_E2 = OFF_E1 + SZ_E;
    const size_t OFF_SIM = 0;                         // aliases e1/e2 (dead)
    size_t off = OFF_E2 + SZ_E;
    const size_t OFF_WPROJT = off; off += 262144;     // 256 x Kp panels
    const size_t OFF_FW1T   = off; off += 262144;
    const size_t OFF_FW2T   = off; off += 262144;
    const size_t OFF_GW1T   = off; off += 262144;
    const size_t OFF_GW2T   = off; off += 262144;
    const size_t OFF_HW1T   = off; off += 262144;
    const size_t OFF_HW2T   = off; off += 262144;
    const size_t OFF_X1P    = off; off += (size_t)M1 * Dn * 4;   // also v1 later
    const size_t OFF_X2P    = off; off += (size_t)M1 * Dn * 4;   // also v2 later
    const size_t OFF_X1PBF  = off; off += (size_t)M1 * DP * 2;
    const size_t OFF_X2PBF  = off; off += (size_t)M1 * DP * 2;
    const size_t OFF_HTMP   = off; off += (size_t)M1 * DP * 2;
    const size_t OFF_F1BF   = off; off += (size_t)M1 * DP * 2;
    const size_t OFF_F2BF   = off; off += (size_t)M1 * DP * 2;
    const size_t OFF_X1PT   = off; off += (size_t)Bn * 256 * Sn * 2;  // 256 pad rows
    const size_t OFF_X2PT   = off; off += (size_t)Bn * 256 * Sn * 2;
    const size_t OFF_PBETA  = off; off += (size_t)Bn * Sn * Sn * 2;
    const size_t OFF_PALPT  = off; off += (size_t)Bn * Sn * Sn * 2;
    const size_t OFF_CAT1   = off; off += (size_t)M1 * D2P * 2;
    const size_t OFF_CAT2   = off; off += (size_t)M1 * D2P * 2;
    const size_t OFF_CATH   = off; off += (size_t)Bn * D2P * 2;
    const size_t OFF_HHTMP  = off; off += (size_t)Bn * DP * 2;
    const size_t OFF_HFIN   = off; off += (size_t)Bn * Dn * 4;

    __bf16* e1    = (__bf16*)(ws + OFF_E1);
    __bf16* e2    = (__bf16*)(ws + OFF_E2);
    float*  sim   = (float*) (ws + OFF_SIM);
    __bf16* WpT   = (__bf16*)(ws + OFF_WPROJT);
    __bf16* FW1T  = (__bf16*)(ws + OFF_FW1T);
    __bf16* FW2T  = (__bf16*)(ws + OFF_FW2T);
    __bf16* GW1T  = (__bf16*)(ws + OFF_GW1T);
    __bf16* GW2T  = (__bf16*)(ws + OFF_GW2T);
    __bf16* HW1T  = (__bf16*)(ws + OFF_HW1T);
    __bf16* HW2T  = (__bf16*)(ws + OFF_HW2T);
    float*  x1p   = (float*) (ws + OFF_X1P);
    float*  x2p   = (float*) (ws + OFF_X2P);
    float*  v1    = x1p;                       // alias (x1p dead by G layer 2)
    float*  v2    = x2p;
    __bf16* x1pbf = (__bf16*)(ws + OFF_X1PBF);
    __bf16* x2pbf = (__bf16*)(ws + OFF_X2PBF);
    __bf16* htmp  = (__bf16*)(ws + OFF_HTMP);
    __bf16* f1bf  = (__bf16*)(ws + OFF_F1BF);
    __bf16* f2bf  = (__bf16*)(ws + OFF_F2BF);
    __bf16* x1pT  = (__bf16*)(ws + OFF_X1PT);
    __bf16* x2pT  = (__bf16*)(ws + OFF_X2PT);
    __bf16* Pbeta = (__bf16*)(ws + OFF_PBETA);
    __bf16* PalpT = (__bf16*)(ws + OFF_PALPT);
    __bf16* cat1  = (__bf16*)(ws + OFF_CAT1);
    __bf16* cat2  = (__bf16*)(ws + OFF_CAT2);
    __bf16* cath  = (__bf16*)(ws + OFF_CATH);
    __bf16* hhtmp = (__bf16*)(ws + OFF_HHTMP);
    float*  hfin  = (float*) (ws + OFF_HFIN);

    auto zero = [&](__bf16* p, long n) {
        zero_bf<<<dim3((unsigned)((n + 255) / 256)), dim3(256), 0, stream>>>(p, n);
    };
    auto gemm = [&](const __bf16* A, int lda, long sA,
                    const __bf16* Bt, int ldb, long sB,
                    float* Cf, int ldc, long sC,
                    __bf16* Cb, int ldcb, long sCb,
                    const float* bias, int M, int N, int K, int relu, int batch) {
        dim3 g((unsigned)((N + 127) / 128), (unsigned)((M + 127) / 128), (unsigned)batch);
        gemm_wmma<<<g, dim3(256), 0, stream>>>(A, lda, sA, Bt, ldb, sB,
                                               Cf, ldc, sC, Cb, ldcb, sCb,
                                               bias, M, N, K, relu);
    };
    auto gemmL = [&](const __bf16* A, int lda, const __bf16* Bt,
                     float* Cf, int ldc, __bf16* Cb, int ldcb,
                     const float* bias, int M, int N, int relu) {
        dim3 g((unsigned)((N + 127) / 128), (unsigned)((M + 127) / 128), 1);
        gemm_wmma_wlds<<<g, dim3(256), 0, stream>>>(A, lda, Bt, Cf, ldc, Cb, ldcb,
                                                    bias, M, N, relu);
    };

    // 0) zero-fill padded bf16 operand buffers
    zero(x1pbf, M1 * DP);  zero(x2pbf, M1 * DP);  zero(htmp, M1 * DP);
    zero(f1bf,  M1 * DP);  zero(f2bf,  M1 * DP);
    zero(cat1,  M1 * D2P); zero(cat2,  M1 * D2P);
    zero(cath,  Bn * D2P); zero(hhtmp, Bn * DP);

    // 1) weights: transpose + convert into 256-row zero-padded panels
    auto twl = [&](const float* W, __bf16* Wt, int K, int Nsrc, int Kp) {
        long n = 256L * Kp;
        transpose_w<<<dim3((unsigned)((n + 255) / 256)), dim3(256), 0, stream>>>(W, Wt, K, Nsrc, Kp);
    };
    twl(W_proj, WpT,  300, 200, EP);
    twl(F_W1,   FW1T, 200, 200, DP);
    twl(F_W2,   FW2T, 200, 200, DP);
    twl(G_W1,   GW1T, 400, 200, D2P);
    twl(G_W2,   GW2T, 200, 200, DP);
    twl(H_W1,   HW1T, 400, 200, D2P);
    twl(H_W2,   HW2T, 200, 200, DP);

    // 2) embed + L2-normalize -> bf16 [M1, 320]
    gather_norm<<<dim3((unsigned)(M1 / 8)), dim3(256), 0, stream>>>(x1, emb, e1, (int)M1);
    gather_norm<<<dim3((unsigned)(M1 / 8)), dim3(256), 0, stream>>>(x2, emb, e2, (int)M1);

    // 3) projection: x{1,2}p = e @ W_proj
    gemm(e1, EP, 0, WpT, EP, 0, x1p, (int)Dn, 0, x1pbf, DP, 0, nullptr, (int)M1, (int)Dn, EP, 0, 1);
    gemm(e2, EP, 0, WpT, EP, 0, x2p, (int)Dn, 0, x2pbf, DP, 0, nullptr, (int)M1, (int)Dn, EP, 0, 1);

    // 4) cat{1,2}[:, 0:200] = x{1,2}p (bf16)
    {
        long n = M1 * Dn;
        copy_slice_bf<<<dim3((unsigned)((n + 255) / 256)), dim3(256), 0, stream>>>(x1pbf, DP, cat1, D2P, M1, (int)Dn);
        copy_slice_bf<<<dim3((unsigned)((n + 255) / 256)), dim3(256), 0, stream>>>(x2pbf, DP, cat2, D2P, M1, (int)Dn);
    }

    // 5) F MLP (two layers, relu) -> f{1,2}bf   [TDM/LDS weight panels]
    gemmL(x1pbf, DP, FW1T, nullptr, 0, htmp, DP, F_b1, (int)M1, (int)Dn, 1);
    gemmL(htmp,  DP, FW2T, nullptr, 0, f1bf, DP, F_b2, (int)M1, (int)Dn, 1);
    gemmL(x2pbf, DP, FW1T, nullptr, 0, htmp, DP, F_b1, (int)M1, (int)Dn, 1);
    gemmL(htmp,  DP, FW2T, nullptr, 0, f2bf, DP, F_b2, (int)M1, (int)Dn, 1);

    // 6) sim[b] = f1[b] @ f2[b]^T   (batched; sim aliases e1/e2 region)
    gemm(f1bf, DP, Sn * DP, f2bf, DP, Sn * DP, sim, (int)Sn, Sn * Sn,
         nullptr, 0, 0, nullptr, (int)Sn, (int)Sn, DP, 0, (int)Bn);

    // 7) multiplicative mask, then row/col softmax -> bf16 probability mats
    {
        long n = Bn * Sn * Sn;
        mask_sim<<<dim3((unsigned)((n + 255) / 256)), dim3(256), 0, stream>>>(sim, len1, len2);
    }
    softmax_rows<<<dim3((unsigned)(M1 / 8)), dim3(256), 0, stream>>>(sim, Pbeta, M1);
    softmax_cols<<<dim3((unsigned)(M1 / 8)), dim3(256), 0, stream>>>(sim, PalpT, M1);

    // 8) transpose projections (256 zero-padded rows): xT[b, d, s]
    {
        long n = Bn * 256 * Sn;
        transpose_x<<<dim3((unsigned)((n + 255) / 256)), dim3(256), 0, stream>>>(x1p, x1pT);
        transpose_x<<<dim3((unsigned)((n + 255) / 256)), dim3(256), 0, stream>>>(x2p, x2pT);
    }

    // 9) beta -> cat1[:,200:400];  alpha -> cat2[:,200:400]   (batched)
    gemm(Pbeta, (int)Sn, Sn * Sn, x2pT, (int)Sn, 256 * Sn, nullptr, 0, 0,
         cat1 + Dn, D2P, Sn * D2P, nullptr, (int)Sn, (int)Dn, (int)Sn, 0, (int)Bn);
    gemm(PalpT, (int)Sn, Sn * Sn, x1pT, (int)Sn, 256 * Sn, nullptr, 0, 0,
         cat2 + Dn, D2P, Sn * D2P, nullptr, (int)Sn, (int)Dn, (int)Sn, 0, (int)Bn);

    // 10) G MLP: layer1 (K=416, direct), layer2 (K=224, TDM/LDS)
    gemm(cat1, D2P, 0, GW1T, D2P, 0, nullptr, 0, 0, htmp, DP, 0, G_b1, (int)M1, (int)Dn, D2P, 1, 1);
    gemmL(htmp, DP, GW2T, v1, (int)Dn, nullptr, 0, G_b2, (int)M1, (int)Dn, 1);
    gemm(cat2, D2P, 0, GW1T, D2P, 0, nullptr, 0, 0, htmp, DP, 0, G_b1, (int)M1, (int)Dn, D2P, 1, 1);
    gemmL(htmp, DP, GW2T, v2, (int)Dn, nullptr, 0, G_b2, (int)M1, (int)Dn, 1);

    // 11) masked aggregate -> cath [128, 416]
    aggregate<<<dim3(200), dim3(256), 0, stream>>>(v1, v2, len1, len2, cath);

    // 12) H MLP -> hfin fp32 [128, 200]
    gemm(cath,  D2P, 0, HW1T, D2P, 0, nullptr, 0, 0, hhtmp, DP, 0, H_b1, (int)Bn, (int)Dn, D2P, 1, 1);
    gemmL(hhtmp, DP, HW2T, hfin, (int)Dn, nullptr, 0, H_b2, (int)Bn, (int)Dn, 1);

    // 13) head: y = hfin @ W_out + b_out   [128, 3]
    out_head<<<dim3(2), dim3(256), 0, stream>>>(hfin, W_out, b_out, y);
}
